// LSTMClassifier_82128364634666
// MI455X (gfx1250) — matrix operand order repaired and checked by reading, not compiled, not run on hardware
//
#include <hip/hip_runtime.h>
#include <hip/hip_bf16.h>

#define VOCAB  50001
#define SEQ    250
#define EMB    100
#define UNITS  128
#define GATES  512      // 4*UNITS
#define DENSE  32
#define NCLS   2
#define BATCH  1024
#define KPAD   128      // EMB padded to WMMA K granularity

typedef __attribute__((ext_vector_type(16))) _Float16 v16h;
typedef __attribute__((ext_vector_type(8)))  _Float16 v8h;
typedef __attribute__((ext_vector_type(8)))  float    v8f;

__device__ __forceinline__ v16h cat16(v8h lo, v8h hi) {
  return __builtin_shufflevector(lo, hi, 0,1,2,3,4,5,6,7,8,9,10,11,12,13,14,15);
}

// A fragment (16x32 f16, MxK). Per CDNA5 ISA 7.12.2:
// lane l holds row m=l&15; halves 0..7 -> K = kc*32 + (l>=16?8:0) + 0..7,
// halves 8..15 -> +16. Two contiguous 16B reads from an LDS row.
__device__ __forceinline__ v16h load_a_frag(const _Float16* rowbase, int kc, int hi8) {
  const _Float16* p = rowbase + kc * 32 + hi8;
  v8h lo = *(const v8h*)p;
  v8h hp = *(const v8h*)(p + 16);
  return cat16(lo, hp);
}

// B fragment (32x16 f16, KxN). lane l holds column n=l&15;
// halves 0..7 -> K = kc*32 + (l>=16?8:0) + 0..7, halves 8..15 -> +16 (matches A).
// Weights are pre-stored column-major [n][KPAD] f16 so this is two contiguous 16B reads.
__device__ __forceinline__ v16h load_b_frag(const _Float16* colbase, int kc, int hi16) {
  const _Float16* p = colbase + kc * 32 + (hi16 >> 1);
  v8h lo = *(const v8h*)p;
  v8h hp = *(const v8h*)(p + 16);
  return cat16(lo, hp);
}

__device__ __forceinline__ float sigm(float x) { return 1.0f / (1.0f + __expf(-x)); }

// ---------------------------------------------------------------------------
// Kernel 0: convert Wx [100,512] f32 -> WxT f16 [512][128] (K padded w/ zeros)
//           and Wh [128,512] f32 -> WhT f16 [512][128], fragment-friendly.
// ---------------------------------------------------------------------------
__global__ void __launch_bounds__(128)
k_prep_weights(const float* __restrict__ Wx, const float* __restrict__ Wh,
               _Float16* __restrict__ WxT, _Float16* __restrict__ WhT) {
  int idx = blockIdx.x * blockDim.x + threadIdx.x;
  if (idx >= GATES * KPAD) return;
  int n = idx >> 7;
  int k = idx & (KPAD - 1);
  WxT[idx] = (k < EMB) ? (_Float16)Wx[(long)k * GATES + n] : (_Float16)0.0f;
  WhT[idx] = (_Float16)Wh[(long)k * GATES + n];
}

// ---------------------------------------------------------------------------
// Kernel 1: fused embedding gather + input projection + LSTM recurrence.
// Block = 256 threads (8 wave32), owns 16 batch rows, loops t = 0..249.
// Wave w computes z columns [64w, 64w+64): z = b + x_t@Wx + h@Wh via WMMA.
// Gates exchanged through LDS; cell state kept f32 in LDS.
// ---------------------------------------------------------------------------
__global__ void __launch_bounds__(256)
k_lstm_fused(const int* __restrict__ tokens, const float* __restrict__ emb,
             const _Float16* __restrict__ WxT, const _Float16* __restrict__ WhT,
             const float* __restrict__ bias, float* __restrict__ hfin) {
  __shared__ _Float16 sX[16][KPAD];        // x_t tile, f16
  __shared__ _Float16 sH[16][KPAD];        // h state, f16 (WMMA A operand)
  __shared__ float    sC[16][UNITS];       // cell state, f32
  __shared__ float    sZ[4][16][UNITS];    // gate pre-activations i,f,g,o

  const int tid  = threadIdx.x;
  const int lane = tid & 31;
  const int wave = tid >> 5;               // 0..7 -> z cols [64*wave, +64)
  const int rowbase = blockIdx.x * 16;     // batch rows [rowbase, rowbase+16)
  const int hi   = lane >> 4;
  const int m16  = lane & 15;
  const int hi8  = hi * 8;
  const int hi16 = hi * 16;

  // h = 0, c = 0
  for (int i = tid; i < 16 * KPAD; i += 256) {
    int r = i >> 7, k = i & 127;
    sH[r][k] = (_Float16)0.0f;
    sC[r][k] = 0.0f;
  }

  // staging map: thread covers sX row tid>>4, 8 columns starting (tid&15)*8
  const int srow = tid >> 4;
  const int scol = (tid & 15) * 8;
  const int* tokrow = tokens + (long)(rowbase + srow) * SEQ;

  __syncthreads();

  for (int t = 0; t < SEQ; ++t) {
    // ---- stage embedding rows for this timestep (f32 -> f16, zero pad) ----
    {
      int tok = tokrow[t];
      const float* erow = emb + (long)tok * EMB;
      #pragma unroll
      for (int j = 0; j < 8; ++j) {
        int k = scol + j;
        sX[srow][k] = (k < EMB) ? (_Float16)erow[k] : (_Float16)0.0f;
      }
      if (t + 1 < SEQ) {  // hint next timestep's embedding row into cache
        __builtin_prefetch(emb + (long)tokrow[t + 1] * EMB, 0, 0);
      }
    }
    __syncthreads();

    // ---- A fragments for x_t and h (reused across this wave's n-tiles) ----
    v16h ax[4], ah[4];
    #pragma unroll
    for (int kc = 0; kc < 4; ++kc) {
      ax[kc] = load_a_frag(&sX[m16][0], kc, hi8);
      ah[kc] = load_a_frag(&sH[m16][0], kc, hi8);
    }

    // ---- z = bias + x@Wx + h@Wh over 4 n-tiles of this wave's slab ----
    #pragma unroll
    for (int nt = 0; nt < 4; ++nt) {
      const int col  = wave * 64 + nt * 16;   // global z column of lane's n
      const float bv = bias[col + m16];
      v8f acc = { bv, bv, bv, bv, bv, bv, bv, bv };
      const _Float16* bxcol = WxT + (long)(col + m16) * KPAD;
      const _Float16* bhcol = WhT + (long)(col + m16) * KPAD;
      #pragma unroll
      for (int kc = 0; kc < 4; ++kc) {
        v16h bx = load_b_frag(bxcol, kc, hi16);
        acc = __builtin_amdgcn_wmma_f32_16x16x32_f16(false, ax[kc], false, bx,
                                                     (short)0, acc, false, false);
      }
      #pragma unroll
      for (int kc = 0; kc < 4; ++kc) {
        v16h bh = load_b_frag(bhcol, kc, hi16);
        acc = __builtin_amdgcn_wmma_f32_16x16x32_f16(false, ah[kc], false, bh,
                                                     (short)0, acc, false, false);
      }
      // scatter accumulator: VGPR r -> row m = r + hi8, gate = col>>7, u = col&127
      const int gcol = col + m16;
      const int gate = gcol >> 7;
      const int u    = gcol & 127;
      #pragma unroll
      for (int r = 0; r < 8; ++r)
        sZ[gate][r + hi8][u] = acc[r];
    }
    __syncthreads();

    // ---- elementwise gate math: c = f*c + i*tanh(g); h = o*tanh(c) ----
    for (int i = tid; i < 16 * UNITS; i += 256) {
      int r = i >> 7, u = i & 127;
      float ig = sigm(sZ[0][r][u]);
      float fg = sigm(sZ[1][r][u]);
      float gg = tanhf(sZ[2][r][u]);
      float og = sigm(sZ[3][r][u]);
      float c  = fg * sC[r][u] + ig * gg;
      float hh = og * tanhf(c);
      sC[r][u] = c;
      sH[r][u] = (_Float16)hh;
      if (t == SEQ - 1)
        hfin[(long)(rowbase + r) * UNITS + u] = hh;
    }
    __syncthreads();
  }
}

// ---------------------------------------------------------------------------
// Kernel 2: dense head. relu(h@W1+b1) @ W2 + b2 -> softmax. One thread/row.
// ~8 MFLOP total; W1/W2 reads are uniform across the wave (L2/K$ friendly).
// ---------------------------------------------------------------------------
__global__ void __launch_bounds__(128)
k_head(const float* __restrict__ hfin, const float* __restrict__ W1,
       const float* __restrict__ b1, const float* __restrict__ W2,
       const float* __restrict__ b2, float* __restrict__ out) {
  int row = blockIdx.x * blockDim.x + threadIdx.x;
  if (row >= BATCH) return;
  const float* h = hfin + (long)row * UNITS;
  float d[DENSE];
  #pragma unroll 4
  for (int j = 0; j < DENSE; ++j) {
    float acc = b1[j];
    for (int k = 0; k < UNITS; ++k) acc = fmaf(h[k], W1[k * DENSE + j], acc);
    d[j] = fmaxf(acc, 0.0f);
  }
  float z0 = b2[0], z1 = b2[1];
  #pragma unroll
  for (int j = 0; j < DENSE; ++j) {
    z0 = fmaf(d[j], W2[j * NCLS + 0], z0);
    z1 = fmaf(d[j], W2[j * NCLS + 1], z1);
  }
  float mx = fmaxf(z0, z1);
  float e0 = __expf(z0 - mx), e1 = __expf(z1 - mx);
  float inv = 1.0f / (e0 + e1);
  out[row * NCLS + 0] = e0 * inv;
  out[row * NCLS + 1] = e1 * inv;
}

extern "C" void kernel_launch(void* const* d_in, const int* in_sizes, int n_in,
                              void* d_out, int out_size, void* d_ws, size_t ws_size,
                              hipStream_t stream) {
  const int*   tokens = (const int*)  d_in[0];
  const float* emb    = (const float*)d_in[1];
  const float* Wx     = (const float*)d_in[2];
  const float* Wh     = (const float*)d_in[3];
  const float* b      = (const float*)d_in[4];
  const float* W1     = (const float*)d_in[5];
  const float* b1     = (const float*)d_in[6];
  const float* W2     = (const float*)d_in[7];
  const float* b2     = (const float*)d_in[8];
  float* out = (float*)d_out;

  // workspace: WxT (128KB f16) | WhT (128KB f16) | hfin (512KB f32) = 768KB
  char* ws = (char*)d_ws;
  _Float16* WxT = (_Float16*)(ws);
  _Float16* WhT = (_Float16*)(ws + (size_t)GATES * KPAD * sizeof(_Float16));
  float*    hfin = (float*)(ws + (size_t)2 * GATES * KPAD * sizeof(_Float16));

  k_prep_weights<<<(GATES * KPAD + 127) / 128, 128, 0, stream>>>(Wx, Wh, WxT, WhT);
  k_lstm_fused<<<BATCH / 16, 256, 0, stream>>>(tokens, emb, WxT, WhT, b, hfin);
  k_head<<<BATCH / 128, 128, 0, stream>>>(hfin, W1, b1, W2, b2, out);
}
